// GDEncoder_50431505989919
// MI455X (gfx1250) — matrix operand order repaired and checked by reading, not compiled, not run on hardware
//
#include <hip/hip_runtime.h>
#include <hip/hip_bf16.h>

typedef __attribute__((ext_vector_type(16))) __bf16 v16bf;
typedef __attribute__((ext_vector_type(8)))  __bf16 v8bf;
typedef __attribute__((ext_vector_type(8)))  float  v8f;

#define Bq  8
#define FHh 8
#define Nn  2048
#define TIN 5
#define H1  32
#define H2  64

__device__ __forceinline__ float eluf(float x) { return x > 0.f ? x : __expf(x) - 1.f; }
__device__ __forceinline__ float sigf(float x) { return 1.f / (1.f + __expf(-x)); }
__device__ __forceinline__ float lreluf(float x) { return fmaxf(x, 0.2f * x); }  // branchless

// ---------------- Phase 1: h1 = ELU(x @ W1^T + b1) ----------------
// x[b,n,k] = hist[b, fh, n, t], k = t*FH + fh
__global__ void k1_traj(const float* __restrict__ hist, const float* __restrict__ W1,
                        const float* __restrict__ b1, float* __restrict__ h1) {
  int tid = blockIdx.x * blockDim.x + threadIdx.x;     // B*N*H1 = 8*2048*32
  int h = tid & 31;
  int n = (tid >> 5) & 2047;
  int b = tid >> 16;
  float acc = b1[h];
#pragma unroll
  for (int t = 0; t < TIN; ++t)
#pragma unroll
    for (int f = 0; f < FHh; ++f) {
      int k = t * FHh + f;
      acc += hist[(((size_t)(b * FHh + f) * Nn) + n) * TIN + t] * W1[h * (TIN * FHh) + k];
    }
  h1[((size_t)b * Nn + n) * H1 + h] = eluf(acc);
}

// ---------------- Phase 2: LSTM over T=8, batch 2048 ----------------
// block = 256 thr = 4 rows x 64 hidden units; h in LDS, c in registers
__global__ void k2_lstm(const float* __restrict__ x, const float* __restrict__ Wih,
                        const float* __restrict__ Whh, const float* __restrict__ bih,
                        const float* __restrict__ bhh, float* __restrict__ hs) {
  __shared__ float hS[4][H2];
  const int j = threadIdx.x & 63;
  const int r = threadIdx.x >> 6;
  const int n = blockIdx.x * 4 + r;
  float c = 0.f;
  hS[r][j] = 0.f;
  const float bi = bih[0 * H2 + j] + bhh[0 * H2 + j];
  const float bf = bih[1 * H2 + j] + bhh[1 * H2 + j];
  const float bg = bih[2 * H2 + j] + bhh[2 * H2 + j];
  const float bo = bih[3 * H2 + j] + bhh[3 * H2 + j];
  __syncthreads();
  for (int t = 0; t < Bq; ++t) {
    float a0 = bi, a1 = bf, a2 = bg, a3 = bo;
    const float* xr = x + ((size_t)t * Nn + n) * H1;
#pragma unroll 4
    for (int k = 0; k < H1; ++k) {
      float xk = xr[k];
      a0 += xk * Wih[(0 * H2 + j) * H1 + k];
      a1 += xk * Wih[(1 * H2 + j) * H1 + k];
      a2 += xk * Wih[(2 * H2 + j) * H1 + k];
      a3 += xk * Wih[(3 * H2 + j) * H1 + k];
    }
#pragma unroll 4
    for (int k = 0; k < H2; ++k) {
      float hk = hS[r][k];
      a0 += hk * Whh[(0 * H2 + j) * H2 + k];
      a1 += hk * Whh[(1 * H2 + j) * H2 + k];
      a2 += hk * Whh[(2 * H2 + j) * H2 + k];
      a3 += hk * Whh[(3 * H2 + j) * H2 + k];
    }
    c = sigf(a1) * c + sigf(a0) * tanhf(a2);
    float h = sigf(a3) * tanhf(c);
    __syncthreads();
    hS[r][j] = h;
    hs[((size_t)t * Nn + n) * H2 + j] = h;
    __syncthreads();
  }
}

// ---------------- Phase 3: Wh = hs@Wg ; WhT(bf16), s, d ----------------
__global__ void k3_wh(const float* __restrict__ hs, const float* __restrict__ Wg,
                      const float* __restrict__ asrc, const float* __restrict__ adst,
                      __bf16* __restrict__ whT, float* __restrict__ s,
                      float* __restrict__ d) {
  __shared__ float red[H2];
  const int r = blockIdx.x;            // 0..16383  (b*2048 + n)
  const int h = threadIdx.x;           // 0..63
  const int b = r >> 11, n = r & 2047;
  float acc = 0.f;
  const float* hr = hs + (size_t)r * H2;
#pragma unroll 8
  for (int k = 0; k < H2; ++k) acc += hr[k] * Wg[k * H2 + h];
  whT[((size_t)(b * H2 + h)) * Nn + n] = (__bf16)acc;

  red[h] = acc * asrc[h];
  for (int off = 32; off; off >>= 1) { __syncthreads(); if (h < off) red[h] += red[h + off]; }
  __syncthreads();
  if (!h) s[r] = red[0];
  __syncthreads();
  red[h] = acc * adst[h];
  for (int off = 32; off; off >>= 1) { __syncthreads(); if (h < off) red[h] += red[h + off]; }
  __syncthreads();
  if (!h) d[r] = red[0];
}

// ---------------- Phase 3b: d_max per batch ----------------
__global__ void k3b_dmax(const float* __restrict__ d, float* __restrict__ dmax) {
  __shared__ float red[256];
  const int b = blockIdx.x;
  float m = -1e30f;
  for (int jj = threadIdx.x; jj < Nn; jj += 256) m = fmaxf(m, d[b * Nn + jj]);
  red[threadIdx.x] = m;
  for (int off = 128; off; off >>= 1) {
    __syncthreads();
    if ((int)threadIdx.x < off) red[threadIdx.x] = fmaxf(red[threadIdx.x], red[threadIdx.x + off]);
  }
  __syncthreads();
  if (!threadIdx.x) dmax[b] = red[0];
}

// ---------------- Phase 4: fused masked-softmax attention via WMMA ----------------
// 1 block = (batch b, 16-row tile). 8 waves stream j-chunks of 32.
// p_ij = mask * exp(lrelu(s_i + d_j) - m_i),  m_i = lrelu(s_i + max_j d_j)  (valid shift)
// out_i = ELU( (P @ Wh)_i / l_i )
__global__ void __launch_bounds__(256, 2)
k4_gat(const int* __restrict__ adj, const float* __restrict__ s,
       const float* __restrict__ d, const float* __restrict__ dmax,
       const __bf16* __restrict__ whT, float* __restrict__ out) {
  __shared__ float accS[16 * H2];
  __shared__ float lS[16];
  const int b  = blockIdx.x >> 7;
  const int i0 = (blockIdx.x & 127) * 16;
  const int tid = threadIdx.x;
  for (int idx = tid; idx < 16 * H2; idx += 256) accS[idx] = 0.f;
  if (tid < 16) lS[tid] = 0.f;
  __syncthreads();

  const int w = tid >> 5, lane = tid & 31;
  const int m = lane & 15, g = lane >> 4;               // A row / K-half per ISA layout
  const int* adjRow = adj + ((size_t)(b * 2 + 1) * Nn + (i0 + m)) * Nn;
  const float* dB = d + b * Nn;
  const __bf16* whB = whT + (size_t)b * H2 * Nn;
  const float sm = s[b * Nn + i0 + m];
  const float mi = lreluf(sm + dmax[b]);                // row-wise softmax shift (upper bound)

  v8f c0 = {}, c1 = {}, c2 = {}, c3 = {};
  float ls0 = 0.f, ls1 = 0.f, ls2 = 0.f, ls3 = 0.f;     // 4 independent chains
  const int kb0 = 8 * g;

  // uniform trip count -> scalar loop control (no exec-mask loop exit)
#pragma unroll 1
  for (int it = 0; it < Nn / 256; ++it) {
    const int jt = w * 32 + it * 256;
    // branchless in-bounds prefetch of next adj tile (wraps within same row)
    __builtin_prefetch(adjRow + ((jt + 256) & (Nn - 1)), 0, 1);  // global_prefetch_b8
    v16bf a;
    float lacc[4];
#pragma unroll
    for (int cc = 0; cc < 4; ++cc) {
      const int kb = ((cc & 1) ? 4 : 0) + ((cc & 2) ? 16 : 0) + kb0;  // {0,4,16,20}+8g
      const int4   mk = *reinterpret_cast<const int4*>(adjRow + jt + kb);
      const float4 dv = *reinterpret_cast<const float4*>(dB + jt + kb);
      const int   mka[4] = {mk.x, mk.y, mk.z, mk.w};
      const float dva[4] = {dv.x, dv.y, dv.z, dv.w};
      float la = 0.f;
#pragma unroll
      for (int e = 0; e < 4; ++e) {
        const float ev  = lreluf(sm + dva[e]);          // v_max_num_f32, no branch
        const float pex = __expf(ev - mi);              // always computed -> no EXEC games
        const float msk = (mka[e] > 0) ? 1.0f : 0.0f;   // v_cndmask on constants
        const float p   = pex * msk;
        la += p;
        a[cc * 4 + e] = (__bf16)p;                      // A frag: K = kb + e (+8g via kb0)
      }
      lacc[cc] = la;
    }
    ls0 += lacc[0]; ls1 += lacc[1]; ls2 += lacc[2]; ls3 += lacc[3];
#pragma unroll
    for (int t = 0; t < 4; ++t) {
      // B frag: lane = column n (t*16+m), K = {0..7}+8g and {16..23}+8g along WhT row
      const __bf16* wr = whB + (size_t)(t * 16 + m) * Nn + jt + 8 * g;
      v8bf lo = *reinterpret_cast<const v8bf*>(wr);
      v8bf hi = *reinterpret_cast<const v8bf*>(wr + 16);
      v16bf bb;
#pragma unroll
      for (int e = 0; e < 8; ++e) { bb[e] = lo[e]; bb[e + 8] = hi[e]; }
      v8f& cacc = (t == 0) ? c0 : (t == 1) ? c1 : (t == 2) ? c2 : c3;
      cacc = __builtin_amdgcn_wmma_f32_16x16x32_bf16(false, a, false, bb,
                                                     (short)0, cacc, false, false);
    }
  }

  atomicAdd(&lS[m], (ls0 + ls1) + (ls2 + ls3));          // ds_add_f32
#pragma unroll
  for (int t = 0; t < 4; ++t) {
    const v8f cv = (t == 0) ? c0 : (t == 1) ? c1 : (t == 2) ? c2 : c3;
#pragma unroll
    for (int r = 0; r < 8; ++r)                          // C: M = r + 8g, N = t*16 + m
      atomicAdd(&accS[(r + 8 * g) * H2 + t * 16 + m], cv[r]);
  }
  __syncthreads();

  for (int idx = tid; idx < 16 * H2; idx += 256) {
    const int mr = idx >> 6, h = idx & 63;
    const float v = accS[idx] / lS[mr];
    out[((size_t)(b * Nn + i0 + mr)) * H2 + h] = eluf(v);
  }
}

// ---------------- launch ----------------
extern "C" void kernel_launch(void* const* d_in, const int* in_sizes, int n_in,
                              void* d_out, int out_size, void* d_ws, size_t ws_size,
                              hipStream_t stream) {
  const float* hist = (const float*)d_in[0];
  const int*   adj  = (const int*)d_in[1];
  const float* W1   = (const float*)d_in[2];
  const float* b1   = (const float*)d_in[3];
  const float* Wih  = (const float*)d_in[4];
  const float* Whh  = (const float*)d_in[5];
  const float* bih  = (const float*)d_in[6];
  const float* bhh  = (const float*)d_in[7];
  const float* Wg   = (const float*)d_in[8];
  const float* asrc = (const float*)d_in[9];
  const float* adst = (const float*)d_in[10];
  float* out = (float*)d_out;
  (void)in_sizes; (void)n_in; (void)out_size; (void)ws_size;

  char* ws = (char*)d_ws;
  float*  h1v  = (float*)(ws);                                   // 2 MB
  float*  hsv  = (float*)(ws + (2ull << 20));                    // 4 MB
  __bf16* whT  = (__bf16*)(ws + (6ull << 20));                   // 2 MB
  float*  sA   = (float*)(ws + (8ull << 20));                    // 64 KB
  float*  dA   = (float*)(ws + (8ull << 20) + 65536);            // 64 KB
  float*  dmax = (float*)(ws + (8ull << 20) + 131072);           // 32 B

  k1_traj<<<2048, 256, 0, stream>>>(hist, W1, b1, h1v);
  k2_lstm<<<512, 256, 0, stream>>>(h1v, Wih, Whh, bih, bhh, hsv);
  k3_wh<<<16384, 64, 0, stream>>>(hsv, Wg, asrc, adst, whT, sA, dA);
  k3b_dmax<<<8, 256, 0, stream>>>(dA, dmax);
  k4_gat<<<1024, 256, 0, stream>>>(adj, sA, dA, dmax, whT, out);
}
